// Ernie4Moe_66881230733995
// MI455X (gfx1250) — matrix operand order repaired and checked by reading, not compiled, not run on hardware
//
#include <hip/hip_runtime.h>
#include <stdint.h>

#define H_  1024
#define E_  16
#define K_  4
#define I_  512
#define SI_ 1024
#define T_  8192
#define TM_ 32      // tokens per block
#define KQ_ 256     // K-quarter staged into LDS at a time

typedef __attribute__((ext_vector_type(8)))  __bf16 v8bf;
typedef __attribute__((ext_vector_type(16))) __bf16 v16bf;
typedef __attribute__((ext_vector_type(8)))  float  v8f;

__device__ __forceinline__ unsigned short f32_to_bf16(float f) {
  union { float f; unsigned int u; } c; c.f = f;
  unsigned int u = c.u;
  u += 0x7FFFu + ((u >> 16) & 1u);   // round-to-nearest-even
  return (unsigned short)(u >> 16);
}

__device__ __forceinline__ v16bf cat16(v8bf lo, v8bf hi) {
  return __builtin_shufflevector(lo, hi, 0,1,2,3,4,5,6,7,8,9,10,11,12,13,14,15);
}

// CDNA5 async global->LDS copy, 16B per lane, tracked with ASYNCcnt.
__device__ __forceinline__ void async_copy_b128(uint32_t lds_off,
                                                const unsigned short* gptr) {
  asm volatile("global_load_async_to_lds_b128 %0, %1, off"
               :: "v"(lds_off), "v"((unsigned long long)(uintptr_t)gptr)
               : "memory");
}
__device__ __forceinline__ void wait_async0() {
  asm volatile("s_wait_asynccnt 0" ::: "memory");
}

// ---------------------------------------------------------------------------
// f32 -> bf16 bulk conversion (x4)
// ---------------------------------------------------------------------------
__global__ void cvt_f32_bf16_kernel(const float* __restrict__ src,
                                    unsigned short* __restrict__ dst, int n4) {
  int i = blockIdx.x * blockDim.x + threadIdx.x;
  if (i >= n4) return;
  float4 v = reinterpret_cast<const float4*>(src)[i];
  ushort4 o;
  o.x = f32_to_bf16(v.x); o.y = f32_to_bf16(v.y);
  o.z = f32_to_bf16(v.z); o.w = f32_to_bf16(v.w);
  reinterpret_cast<ushort4*>(dst)[i] = o;
}

__global__ void zero_counts_kernel(int* cnt) {
  if (threadIdx.x < E_) cnt[threadIdx.x] = 0;
}

__global__ void zero_out_kernel(float* __restrict__ out, int n4) {
  int i = blockIdx.x * blockDim.x + threadIdx.x;
  if (i >= n4) return;
  reinterpret_cast<float4*>(out)[i] = float4{0.f, 0.f, 0.f, 0.f};
}

// ---------------------------------------------------------------------------
// Router: sigmoid gating, top-4 on bias-corrected scores, renormalized
// combine weights from uncorrected scores; scatter into per-expert lists.
// ---------------------------------------------------------------------------
__global__ void gate_kernel(const float* __restrict__ x,
                            const float* __restrict__ gw,
                            const float* __restrict__ gb,
                            int* __restrict__ cnt,
                            int* __restrict__ tokIdx,
                            float* __restrict__ tokW) {
  int t = blockIdx.x * blockDim.x + threadIdx.x;
  if (t >= T_) return;
  float acc[E_];
#pragma unroll
  for (int e = 0; e < E_; ++e) acc[e] = 0.f;
  const float4* xr = reinterpret_cast<const float4*>(x + (size_t)t * H_);
  for (int h = 0; h < H_ / 4; ++h) {
    float4 xv = xr[h];
#pragma unroll
    for (int e = 0; e < E_; ++e) {
      float4 wv = reinterpret_cast<const float4*>(gw + (size_t)e * H_)[h];
      acc[e] += xv.x * wv.x + xv.y * wv.y + xv.z * wv.z + xv.w * wv.w;
    }
  }
  float score[E_], corr[E_];
#pragma unroll
  for (int e = 0; e < E_; ++e) {
    score[e] = 1.f / (1.f + __expf(-acc[e]));
    corr[e]  = score[e] + gb[e];
  }
  int ti[K_]; float tw[K_]; float s = 0.f;
#pragma unroll
  for (int k = 0; k < K_; ++k) {
    int best = 0; float bv = -1e30f;
#pragma unroll
    for (int e = 0; e < E_; ++e)
      if (corr[e] > bv) { bv = corr[e]; best = e; }
    corr[best] = -1e30f;
    ti[k] = best; tw[k] = score[best]; s += score[best];
  }
  float inv = 1.f / s;
#pragma unroll
  for (int k = 0; k < K_; ++k) {
    int e = ti[k];
    int pos = atomicAdd(&cnt[e], 1);
    tokIdx[(size_t)e * T_ + pos] = t;
    tokW[(size_t)e * T_ + pos]  = tw[k] * inv;
  }
}

// ---------------------------------------------------------------------------
// Unified fused SwiGLU MLP (I = 512 per instance):
//   blockIdx.y < 16 : routed expert e, gathered token list, scale = tokW
//   blockIdx.y >= 16: shared-expert sub-slab s = y-16 (SI=1024 split in two
//                     I=512 slabs along the intermediate dim), dense tokens,
//                     scale = 1
// 32 tokens/block, 8 waves. Per wave: 2 M-tiles; B fragments reused across
// both (2x arithmetic intensity vs 1 M-tile). x tile staged to LDS in
// K-quarters with global_load_async_to_lds_b128 + s_wait_asynccnt.
// Output accumulated with global atomic f32 adds into zeroed out.
// ---------------------------------------------------------------------------
__launch_bounds__(256)
__global__ void moe_mlp_kernel(const unsigned short* __restrict__ xb,
                               const unsigned short* __restrict__ w13b,   // [E][2I][H]
                               const unsigned short* __restrict__ w2b,    // [E][H][I]
                               const unsigned short* __restrict__ sw13b,  // [2SI][H]
                               const unsigned short* __restrict__ sw2b,   // [H][SI]
                               float* __restrict__ out,
                               const int* __restrict__ cnt,
                               const int* __restrict__ tokIdx,
                               const float* __restrict__ tokW) {
  __shared__ unsigned short xLds[TM_ * KQ_];   // 16 KB
  __shared__ unsigned short aLds[TM_ * I_];    // 32 KB
  __shared__ int   toks[TM_];
  __shared__ float scales[TM_];

  const int  e     = blockIdx.y;
  const bool dense = (e >= E_);
  const int  tile  = blockIdx.x;

  const unsigned short* w13e;
  const unsigned short* w2e;
  int voff, ldw2, c;
  if (dense) {
    const int s = e - E_;
    w13e = sw13b + (size_t)s * I_ * H_;   // u rows [s*512, s*512+512)
    voff = SI_;                           // v rows at +SI in sw13
    w2e  = sw2b + (size_t)s * I_;         // column slab of sw2
    ldw2 = SI_;
    c    = T_;
  } else {
    c = cnt[e];
    if (tile * TM_ >= c) return;          // uniform per block
    w13e = w13b + (size_t)e * (2 * I_) * H_;
    voff = I_;
    w2e  = w2b + (size_t)e * H_ * I_;
    ldw2 = I_;
  }

  if (threadIdx.x < TM_) {
    int gi = tile * TM_ + threadIdx.x;
    if (dense) {
      toks[threadIdx.x]   = gi;
      scales[threadIdx.x] = 1.f;
    } else {
      int src = (gi < c) ? gi : (c - 1);
      toks[threadIdx.x]   = tokIdx[(size_t)e * T_ + src];
      scales[threadIdx.x] = (gi < c) ? tokW[(size_t)e * T_ + src] : 0.f;
    }
  }

  // Staging plan: 32 rows x 256 cols bf16 = 1024 x 16B segments; 4/thread.
  const unsigned short* srcRow[4];
  uint32_t ldsDst[4];
  {
    uint32_t xbase = (uint32_t)(uintptr_t)&xLds[0];
#pragma unroll
    for (int r = 0; r < 4; ++r) {
      int seg = threadIdx.x + 256 * r;   // 0..1023
      int row = seg >> 5;                // 32 segments per row
      int cs  = seg & 31;
      int gi  = tile * TM_ + row;
      int tok;
      if (dense) tok = gi;
      else { int src = (gi < c) ? gi : (c - 1); tok = tokIdx[(size_t)e * T_ + src]; }
      srcRow[r] = xb + (size_t)tok * H_ + cs * 8;
      ldsDst[r] = xbase + (uint32_t)(row * KQ_ + cs * 8) * 2u;
    }
  }

  const int lane    = threadIdx.x & 31;
  const int wave    = threadIdx.x >> 5;
  const int halfSel = lane >> 4;   // 0: lanes 0-15, 1: lanes 16-31
  const int n0      = lane & 15;

  // ---- Phase 1: u/v projections + SwiGLU -> aLds ----
  for (int chunk = 0; chunk < 2; ++chunk) {
    const int jbase = wave * 64 + chunk * 32;   // 32 j-columns per wave/chunk
    v8f cu[2][2] = {}; v8f cv[2][2] = {};       // [mt][nt]
    for (int q = 0; q < 4; ++q) {
      __syncthreads();                          // prior xLds readers done
#pragma unroll
      for (int r = 0; r < 4; ++r)
        async_copy_b128(ldsDst[r], srcRow[r] + q * KQ_);
      wait_async0();
      __syncthreads();                          // xLds visible to all waves

      const int kg0 = q * KQ_;
      for (int ksq = 0; ksq < KQ_; ksq += 32) {
        v16bf af[2];
#pragma unroll
        for (int mt = 0; mt < 2; ++mt) {
          const unsigned short* pa =
              &xLds[(mt * 16 + n0) * KQ_ + ksq + (halfSel << 4)];
          af[mt] = cat16(*reinterpret_cast<const v8bf*>(pa),
                         *reinterpret_cast<const v8bf*>(pa + 8));
        }
        const int kg = kg0 + ksq;
#pragma unroll
        for (int nt = 0; nt < 2; ++nt) {
          int j = jbase + nt * 16 + n0;
          const unsigned short* pu = w13e + (size_t)j * H_ + kg + (halfSel << 4);
          const unsigned short* pv = w13e + (size_t)(voff + j) * H_ + kg + (halfSel << 4);
          v16bf bu = *reinterpret_cast<const v16bf*>(pu);
          v16bf bv = *reinterpret_cast<const v16bf*>(pv);
#pragma unroll
          for (int mt = 0; mt < 2; ++mt) {
            cu[mt][nt] = __builtin_amdgcn_wmma_f32_16x16x32_bf16(
                false, af[mt], false, bu, (short)0, cu[mt][nt], false, false);
            cv[mt][nt] = __builtin_amdgcn_wmma_f32_16x16x32_bf16(
                false, af[mt], false, bv, (short)0, cv[mt][nt], false, false);
          }
        }
      }
    }
    // SwiGLU -> aLds (bf16)
#pragma unroll
    for (int mt = 0; mt < 2; ++mt) {
#pragma unroll
      for (int nt = 0; nt < 2; ++nt) {
        int col = jbase + nt * 16 + n0;
#pragma unroll
        for (int r = 0; r < 8; ++r) {
          int m = mt * 16 + halfSel * 8 + r;
          float u = cu[mt][nt][r], v = cv[mt][nt][r];
          aLds[m * I_ + col] = f32_to_bf16((u / (1.f + __expf(-u))) * v);
        }
      }
    }
  }
  __syncthreads();

  // ---- Phase 2: down projection, 128 H-cols per wave in two 64-col halves ----
  for (int hh = 0; hh < 2; ++hh) {
    const int hbase = wave * 128 + hh * 64;
    v8f cy[2][4] = {};
    for (int ks = 0; ks < I_; ks += 32) {
      v16bf af[2];
#pragma unroll
      for (int mt = 0; mt < 2; ++mt) {
        const unsigned short* pa =
            &aLds[(mt * 16 + n0) * I_ + ks + (halfSel << 4)];
        af[mt] = cat16(*reinterpret_cast<const v8bf*>(pa),
                       *reinterpret_cast<const v8bf*>(pa + 8));
      }
#pragma unroll
      for (int nt = 0; nt < 4; ++nt) {
        int h = hbase + nt * 16 + n0;
        const unsigned short* pb = w2e + (size_t)h * ldw2 + ks + (halfSel << 4);
        v16bf b = *reinterpret_cast<const v16bf*>(pb);
#pragma unroll
        for (int mt = 0; mt < 2; ++mt) {
          cy[mt][nt] = __builtin_amdgcn_wmma_f32_16x16x32_bf16(
              false, af[mt], false, b, (short)0, cy[mt][nt], false, false);
        }
      }
    }
#pragma unroll
    for (int mt = 0; mt < 2; ++mt) {
#pragma unroll
      for (int nt = 0; nt < 4; ++nt) {
        int h = hbase + nt * 16 + n0;
#pragma unroll
        for (int r = 0; r < 8; ++r) {
          int m = mt * 16 + halfSel * 8 + r;
          atomicAdd(out + (size_t)toks[m] * H_ + h, scales[m] * cy[mt][nt][r]);
        }
      }
    }
  }
}

// ---------------------------------------------------------------------------
// Launch
// ---------------------------------------------------------------------------
extern "C" void kernel_launch(void* const* d_in, const int* in_sizes, int n_in,
                              void* d_out, int out_size, void* d_ws, size_t ws_size,
                              hipStream_t stream) {
  const float* x    = (const float*)d_in[0];
  const float* gw   = (const float*)d_in[1];
  const float* gb   = (const float*)d_in[2];
  const float* w13  = (const float*)d_in[3];
  const float* w2   = (const float*)d_in[4];
  const float* sw13 = (const float*)d_in[5];
  const float* sw2  = (const float*)d_in[6];
  float* out = (float*)d_out;

  char* ws = (char*)d_ws;
  size_t off = 0;
  auto take = [&](size_t bytes) -> char* {
    char* p = ws + off;
    off = (off + bytes + 255) & ~(size_t)255;
    return p;
  };
  unsigned short* xb    = (unsigned short*)take((size_t)T_ * H_ * 2);
  unsigned short* w13b  = (unsigned short*)take((size_t)E_ * 2 * I_ * H_ * 2);
  unsigned short* w2b   = (unsigned short*)take((size_t)E_ * H_ * I_ * 2);
  unsigned short* sw13b = (unsigned short*)take((size_t)2 * SI_ * H_ * 2);
  unsigned short* sw2b  = (unsigned short*)take((size_t)H_ * SI_ * 2);
  int*            cnt   = (int*)take((size_t)E_ * sizeof(int));
  int*            tokIdx= (int*)take((size_t)E_ * T_ * sizeof(int));
  float*          tokW  = (float*)take((size_t)E_ * T_ * sizeof(float));

  auto cvt = [&](const float* src, unsigned short* dst, size_t n) {
    int n4 = (int)(n / 4);
    cvt_f32_bf16_kernel<<<(n4 + 255) / 256, 256, 0, stream>>>(src, dst, n4);
  };

  zero_counts_kernel<<<1, 32, 0, stream>>>(cnt);
  {
    int n4 = (T_ * H_) / 4;
    zero_out_kernel<<<(n4 + 255) / 256, 256, 0, stream>>>(out, n4);
  }
  cvt(x,    xb,    (size_t)T_ * H_);
  cvt(w13,  w13b,  (size_t)E_ * 2 * I_ * H_);
  cvt(w2,   w2b,   (size_t)E_ * H_ * I_);
  cvt(sw13, sw13b, (size_t)2 * SI_ * H_);
  cvt(sw2,  sw2b,  (size_t)H_ * SI_);

  gate_kernel<<<T_ / 256, 256, 0, stream>>>(x, gw, gb, cnt, tokIdx, tokW);

  // 16 routed experts + 2 shared-expert slabs in one launch.
  moe_mlp_kernel<<<dim3(T_ / TM_, E_ + 2), 256, 0, stream>>>(
      xb, w13b, w2b, sw13b, sw2b, out, cnt, tokIdx, tokW);
}
